// MCFL_49254684950998
// MI455X (gfx1250) — matrix operand order repaired and checked by reading, not compiled, not run on hardware
//
#include <hip/hip_runtime.h>
#include <hip/hip_bf16.h>

typedef __attribute__((ext_vector_type(16))) __bf16 bf16x16;
typedef __attribute__((ext_vector_type(8)))  __bf16 bf16x8;
typedef __attribute__((ext_vector_type(8)))  float  f32x8;

typedef unsigned int u32x4 __attribute__((ext_vector_type(4)));
typedef int          i32x8 __attribute__((ext_vector_type(8)));
typedef int          i32x4 __attribute__((ext_vector_type(4)));

typedef __attribute__((address_space(1))) i32x4 As1V4;   // global v4i
typedef __attribute__((address_space(3))) i32x4 As3V4;   // LDS v4i
typedef __attribute__((address_space(3))) __bf16 LdsBf16;

#if defined(__AMDGCN__) && __has_builtin(__builtin_amdgcn_global_load_async_to_lds_b128)
#define HAVE_ASYNC 1
#else
#define HAVE_ASYNC 0
#endif
#if defined(__AMDGCN__) && __has_builtin(__builtin_amdgcn_tensor_load_to_lds) && \
    __has_builtin(__builtin_amdgcn_s_wait_tensorcnt)
#define HAVE_TDM 1
#else
#define HAVE_TDM 0
#endif

#define NB     16384
#define DDIM   1024
#define NHEAD  16
#define HDIM   64
#define ATTN_SCALE 0.125f
#define LN_EPS 1e-5f

__device__ inline void wait_async0() {
#if HAVE_ASYNC
#if __has_builtin(__builtin_amdgcn_s_wait_asynccnt)
  __builtin_amdgcn_s_wait_asynccnt(0);
#else
  asm volatile("s_wait_asynccnt 0x0" ::: "memory");
#endif
#endif
}

#if HAVE_ASYNC
// one 16-byte async DMA: global -> LDS (ASYNCcnt)
__device__ inline void async_cp16(const void* gsrc, void* ldst) {
  __builtin_amdgcn_global_load_async_to_lds_b128(
      (As1V4*)(void*)gsrc, (As3V4*)ldst, 0, 0);
}
#endif

// ---------------------------------------------------------------------------
// Pack the three modality streams into tokens[B*3, D] as bf16
// ---------------------------------------------------------------------------
__global__ __launch_bounds__(256) void pack_tokens(const float* __restrict__ ct,
                                                   const float* __restrict__ ci,
                                                   const float* __restrict__ ca,
                                                   __bf16* __restrict__ tok) {
  size_t idx = (size_t)blockIdx.x * 256 + threadIdx.x;   // < 49152*1024
  size_t row = idx >> 10;
  int    col = (int)(idx & 1023);
  size_t b = row / 3;
  int    t = (int)(row % 3);
  const float* s = (t == 0) ? ct : (t == 1) ? ci : ca;
  tok[idx] = (__bf16)s[b * 1024 + col];
}

// ---------------------------------------------------------------------------
// W[K][N] fp32 -> Wt[N][K] bf16  (weights are tiny: 8M elements total)
// ---------------------------------------------------------------------------
__global__ __launch_bounds__(256) void transpose_w(const float* __restrict__ W,
                                                   __bf16* __restrict__ Wt,
                                                   int K, int N) {
  size_t idx = (size_t)blockIdx.x * 256 + threadIdx.x;
  if (idx >= (size_t)K * N) return;
  int k = (int)(idx / N);
  int n = (int)(idx % N);
  Wt[(size_t)n * K + k] = (__bf16)W[idx];
}

// ---------------------------------------------------------------------------
// TDM descriptor for one weight tile: 128 rows x 64 bf16, row stride K elems,
// into LDS rows of 72 bf16 (pad 16B after every 128B row => interval code 4,
// amount code 3).  Issued by one wave; completion via TENSORcnt.
// ---------------------------------------------------------------------------
#if HAVE_TDM
__device__ inline void tdm_load_b_tile(const __bf16* gsrc, unsigned lds_addr,
                                       int K, int N) {
  unsigned long long ga = (unsigned long long)(uintptr_t)gsrc;
  u32x4 g0;
  g0[0] = 1u;                                     // count=1 valid user D#
  g0[1] = lds_addr;                               // LDS byte address
  g0[2] = (unsigned)(ga & 0xFFFFFFFFu);           // global_addr[31:0]
  g0[3] = (unsigned)((ga >> 32) & 0x1FFFFFFu) | (2u << 30);  // addr[56:32] | type=2
  i32x8 g1;
  g1[0] = (1 << 16)      // data_size = 2 bytes
        | (1 << 20)      // pad_enable
        | (4 << 22)      // pad_interval: 32 DWORDs (128B) between pads
        | (3 << 25);     // pad_amount: 4 DWORDs (16B)
  g1[1] = (int)((unsigned)K << 16);                       // tensor_dim0 lo16
  g1[2] = (int)((((unsigned)K >> 16) & 0xFFFFu) | ((unsigned)N << 16)); // dim0 hi | dim1 lo
  g1[3] = (int)(((unsigned)N >> 16) & 0xFFFFu) | (64 << 16);            // dim1 hi | tile_dim0=64
  g1[4] = 128;                                            // tile_dim1=128, tile_dim2=0
  g1[5] = K;                                              // tensor_dim0_stride lo32
  g1[6] = 0;
  g1[7] = 0;
  i32x4 gz4 = {0, 0, 0, 0};
  i32x8 gz8 = {0, 0, 0, 0, 0, 0, 0, 0};
  __builtin_amdgcn_tensor_load_to_lds(g0, g1, gz4, gz4, gz8, 0);
}
#endif

// ---------------------------------------------------------------------------
// bf16 WMMA GEMM:  C[M,N] (fp32, +bias) = A[M,K] * Wt[N,K]^T
//   block = 256 threads = 8 wave32; block tile 128x128, BK=64, double-buffered
//   A tile: per-lane GLOBAL_LOAD_ASYNC_TO_LDS_B128 (ASYNCcnt)
//   B tile: one TENSOR_LOAD_TO_LDS from wave 0 (TENSORcnt)
// MODE 0: A row -> element offset row*ldA
// MODE 1: A row -> tokens2 rows skipping token 0 of each triple
// ---------------------------------------------------------------------------
template <int MODE>
__global__ __launch_bounds__(256) void gemm_bf16_wmma(
    const __bf16* __restrict__ A, const __bf16* __restrict__ Bw,
    float* __restrict__ C, const float* __restrict__ bias,
    int M, int N, int K, int ldA) {
  __shared__ __bf16 As[2][128][72];   // 72 = 64 + 8 bf16 pad (keeps rows 16B-aligned)
  __shared__ __bf16 Bs[2][128][72];

  const int tid  = threadIdx.x;
  const int lane = tid & 31;
  const int wave = tid >> 5;
  const int lr   = lane & 15;
  const int half = lane >> 4;
  const int waveM = (wave & 3) * 32;   // 4 row groups
  const int waveN = (wave >> 2) * 64;  // 2 col groups
  const int nBase = blockIdx.x * 128;
  const int mBase = blockIdx.y * 128;

  const f32x8 fzero = {};
  f32x8 acc[2][4];
#pragma unroll
  for (int r = 0; r < 2; ++r)
#pragma unroll
    for (int c = 0; c < 4; ++c) acc[r][c] = fzero;

  auto aRowOff = [&](int r) -> size_t {
    if (MODE == 0) return (size_t)(mBase + r) * (size_t)ldA;
    int row = mBase + r;
    return ((size_t)(row >> 1) * 3 + 1 + (size_t)(row & 1)) * 1024;
  };

  // A tile: 128 rows x 64 bf16 = 1024 16B chunks, 4 per thread
  auto loadA = [&](int s, int kk) {
#pragma unroll
    for (int it = 0; it < 4; ++it) {
      int chunk = tid + it * 256;
      int r  = chunk >> 3;
      int ck = (chunk & 7) << 3;
      size_t aoff = aRowOff(r) + kk + ck;
#if HAVE_ASYNC
      async_cp16(&A[aoff], &As[s][r][ck]);
#else
      *(bf16x8*)&As[s][r][ck] = *(const bf16x8*)&A[aoff];
#endif
    }
  };

  auto loadB = [&](int s, int kk) {
#if HAVE_TDM
    if (wave == 0) {
      unsigned lds_addr = (unsigned)(unsigned long long)(LdsBf16*)&Bs[s][0][0];
      tdm_load_b_tile(&Bw[(size_t)nBase * K + kk], lds_addr, K, N);
    }
#else
#pragma unroll
    for (int it = 0; it < 4; ++it) {
      int chunk = tid + it * 256;
      int r  = chunk >> 3;
      int ck = (chunk & 7) << 3;
      const __bf16* src = &Bw[(size_t)(nBase + r) * K + kk + ck];
#if HAVE_ASYNC
      async_cp16(src, &Bs[s][r][ck]);
#else
      *(bf16x8*)&Bs[s][r][ck] = *(const bf16x8*)src;
#endif
    }
#endif
  };

  auto publishStage = [&]() {
    wait_async0();                              // my wave's A (and maybe B) DMAs
#if HAVE_TDM
    if (wave == 0) __builtin_amdgcn_s_wait_tensorcnt(0);  // weight tile DMA
#endif
    __syncthreads();
  };

  loadA(0, 0);
  loadB(0, 0);
  publishStage();

  for (int kk = 0; kk < K; kk += 64) {
    const int s = (kk >> 6) & 1;
    if (kk + 64 < K) {
      loadA(s ^ 1, kk + 64);
      loadB(s ^ 1, kk + 64);
      if (kk + 128 < K) {  // warm L2 for the stage after next (global_prefetch_b8)
        __builtin_prefetch(&Bw[(size_t)(nBase + (tid >> 1)) * K + kk + 128], 0, 1);
        __builtin_prefetch(&A[aRowOff(tid >> 1) + kk + 128], 0, 1);
      }
    }

    union Frag { bf16x16 v; bf16x8 h[2]; };
#pragma unroll
    for (int kt = 0; kt < 64; kt += 32) {
      Frag af[2], bfr[4];
#pragma unroll
      for (int r = 0; r < 2; ++r) {
        int row = waveM + r * 16 + lr;
        af[r].h[0] = *(const bf16x8*)&As[s][row][kt + half * 8];
        af[r].h[1] = *(const bf16x8*)&As[s][row][kt + 16 + half * 8];
      }
#pragma unroll
      for (int c = 0; c < 4; ++c) {
        int col = waveN + c * 16 + lr;
        bfr[c].h[0] = *(const bf16x8*)&Bs[s][col][kt + half * 8];
        bfr[c].h[1] = *(const bf16x8*)&Bs[s][col][kt + 16 + half * 8];
      }
#pragma unroll
      for (int r = 0; r < 2; ++r)
#pragma unroll
        for (int c = 0; c < 4; ++c)
          acc[r][c] = __builtin_amdgcn_wmma_f32_16x16x32_bf16(
              false, af[r].v, false, bfr[c].v, (short)0, acc[r][c], false, false);
    }
    publishStage();
  }

#pragma unroll
  for (int c = 0; c < 4; ++c) {
    int gn = nBase + waveN + c * 16 + lr;
    float bv = bias ? bias[gn] : 0.f;
#pragma unroll
    for (int r = 0; r < 2; ++r) {
#pragma unroll
      for (int i = 0; i < 8; ++i) {
        int gm = mBase + waveM + r * 16 + half * 8 + i;
        C[(size_t)gm * N + gn] = acc[r][c][i] + bv;
      }
    }
  }
}

// ---------------------------------------------------------------------------
// Self-attention, seq len 3: one wave32 per (batch, head); each lane owns 2 dims
// qkv fp32 [B*3, 3072]; out bf16 [B*3, 1024] (merged heads)
// ---------------------------------------------------------------------------
__global__ __launch_bounds__(256) void attn_self_kernel(const float* __restrict__ qkv,
                                                        __bf16* __restrict__ out) {
  int gw   = blockIdx.x * 8 + (threadIdx.x >> 5);
  int lane = threadIdx.x & 31;
  int b = gw >> 4;
  int h = gw & 15;
  const float* base = qkv + (size_t)b * 3 * 3072 + h * 64 + lane * 2;

  float q[3][2], k[3][2], v[3][2];
#pragma unroll
  for (int i = 0; i < 3; ++i) {
    const float* r = base + (size_t)i * 3072;
    q[i][0] = r[0];    q[i][1] = r[1];
    k[i][0] = r[1024]; k[i][1] = r[1025];
    v[i][0] = r[2048]; v[i][1] = r[2049];
  }
  float s[3][3];
#pragma unroll
  for (int i = 0; i < 3; ++i)
#pragma unroll
    for (int j = 0; j < 3; ++j) {
      float p = q[i][0] * k[j][0] + q[i][1] * k[j][1];
#pragma unroll
      for (int m = 16; m >= 1; m >>= 1) p += __shfl_xor(p, m, 32);
      s[i][j] = p * ATTN_SCALE;
    }
#pragma unroll
  for (int i = 0; i < 3; ++i) {
    float mx = fmaxf(fmaxf(s[i][0], s[i][1]), s[i][2]);
    float e0 = __expf(s[i][0] - mx), e1 = __expf(s[i][1] - mx), e2 = __expf(s[i][2] - mx);
    float inv = 1.f / (e0 + e1 + e2);
#pragma unroll
    for (int d = 0; d < 2; ++d) {
      float o = (e0 * v[0][d] + e1 * v[1][d] + e2 * v[2][d]) * inv;
      out[(size_t)(b * 3 + i) * 1024 + h * 64 + lane * 2 + d] = (__bf16)o;
    }
  }
}

// ---------------------------------------------------------------------------
// Cross-attention, 1 query x 2 keys: one wave32 per (batch, head)
// ---------------------------------------------------------------------------
__global__ __launch_bounds__(256) void attn_cross_kernel(const float* __restrict__ qc,
                                                         const float* __restrict__ kvc,
                                                         __bf16* __restrict__ out) {
  int gw   = blockIdx.x * 8 + (threadIdx.x >> 5);
  int lane = threadIdx.x & 31;
  int b = gw >> 4;
  int h = gw & 15;
  const float* qp = qc + (size_t)b * 1024 + h * 64 + lane * 2;
  float q0 = qp[0], q1 = qp[1];

  float k[2][2], v[2][2], s[2];
#pragma unroll
  for (int t = 0; t < 2; ++t) {
    const float* kp = kvc + (size_t)(b * 2 + t) * 2048 + h * 64 + lane * 2;
    k[t][0] = kp[0];    k[t][1] = kp[1];
    v[t][0] = kp[1024]; v[t][1] = kp[1025];
    float p = q0 * k[t][0] + q1 * k[t][1];
#pragma unroll
    for (int m = 16; m >= 1; m >>= 1) p += __shfl_xor(p, m, 32);
    s[t] = p * ATTN_SCALE;
  }
  float mx = fmaxf(s[0], s[1]);
  float e0 = __expf(s[0] - mx), e1 = __expf(s[1] - mx);
  float inv = 1.f / (e0 + e1);
#pragma unroll
  for (int d = 0; d < 2; ++d) {
    float o = (e0 * v[0][d] + e1 * v[1][d]) * inv;
    out[(size_t)b * 1024 + h * 64 + lane * 2 + d] = (__bf16)o;
  }
}

// ---------------------------------------------------------------------------
// LayerNorm helpers: one 256-thread block per 1024-wide row
// ---------------------------------------------------------------------------
__device__ inline float block_reduce_sum(float v, float* sm) {
#pragma unroll
  for (int m = 16; m >= 1; m >>= 1) v += __shfl_xor(v, m, 32);
  int w = threadIdx.x >> 5, lane = threadIdx.x & 31;
  if (lane == 0) sm[w] = v;
  __syncthreads();
  if (w == 0) {
    float t = (lane < 8) ? sm[lane] : 0.f;
#pragma unroll
    for (int m = 4; m >= 1; m >>= 1) t += __shfl_xor(t, m, 32);
    if (lane == 0) sm[0] = t;
  }
  __syncthreads();
  float r = sm[0];
  __syncthreads();
  return r;
}

// tokens2 = LN(tokens + sa); emit bf16 for next GEMMs, keep token-0 rows fp32
__global__ __launch_bounds__(256) void add_ln1_kernel(
    const float* __restrict__ ct, const float* __restrict__ ci, const float* __restrict__ cau,
    const float* __restrict__ sa, const float* __restrict__ g, const float* __restrict__ bta,
    __bf16* __restrict__ tok2, float* __restrict__ t0) {
  __shared__ float sm[8];
  int row = blockIdx.x;
  int b = row / 3, t = row % 3;
  const float* src = (t == 0) ? ct : (t == 1) ? ci : cau;
  const float* x0 = src + (size_t)b * 1024 + threadIdx.x * 4;
  const float* x1 = sa + (size_t)row * 1024 + threadIdx.x * 4;

  float x[4];
  float sum = 0.f;
#pragma unroll
  for (int j = 0; j < 4; ++j) { x[j] = x0[j] + x1[j]; sum += x[j]; }
  float mean = block_reduce_sum(sum, sm) * (1.f / 1024.f);
  float sq = 0.f;
#pragma unroll
  for (int j = 0; j < 4; ++j) { float d = x[j] - mean; sq += d * d; }
  float var  = block_reduce_sum(sq, sm) * (1.f / 1024.f);
  float rstd = rsqrtf(var + LN_EPS);
  int cb = threadIdx.x * 4;
#pragma unroll
  for (int j = 0; j < 4; ++j) {
    float y = (x[j] - mean) * rstd * g[cb + j] + bta[cb + j];
    tok2[(size_t)row * 1024 + cb + j] = (__bf16)y;
    if (t == 0) t0[(size_t)b * 1024 + cb + j] = y;
  }
}

// out = LN(t0 + ca)
__global__ __launch_bounds__(256) void final_ln_kernel(
    const float* __restrict__ t0, const float* __restrict__ caf,
    const float* __restrict__ g, const float* __restrict__ bta,
    float* __restrict__ out) {
  __shared__ float sm[8];
  int b = blockIdx.x;
  const float* x0 = t0 + (size_t)b * 1024 + threadIdx.x * 4;
  const float* x1 = caf + (size_t)b * 1024 + threadIdx.x * 4;
  float x[4];
  float sum = 0.f;
#pragma unroll
  for (int j = 0; j < 4; ++j) { x[j] = x0[j] + x1[j]; sum += x[j]; }
  float mean = block_reduce_sum(sum, sm) * (1.f / 1024.f);
  float sq = 0.f;
#pragma unroll
  for (int j = 0; j < 4; ++j) { float d = x[j] - mean; sq += d * d; }
  float var  = block_reduce_sum(sq, sm) * (1.f / 1024.f);
  float rstd = rsqrtf(var + LN_EPS);
  int cb = threadIdx.x * 4;
#pragma unroll
  for (int j = 0; j < 4; ++j) {
    float y = (x[j] - mean) * rstd * g[cb + j] + bta[cb + j];
    out[(size_t)b * 1024 + cb + j] = y;
  }
}

// ---------------------------------------------------------------------------
extern "C" void kernel_launch(void* const* d_in, const int* in_sizes, int n_in,
                              void* d_out, int out_size, void* d_ws, size_t ws_size,
                              hipStream_t stream) {
  (void)in_sizes; (void)n_in; (void)out_size; (void)ws_size;
  const float* c_text  = (const float*)d_in[0];
  const float* c_image = (const float*)d_in[1];
  const float* c_audio = (const float*)d_in[2];
  const float* Wqkv    = (const float*)d_in[3];
  const float* sa_w    = (const float*)d_in[4];
  const float* sa_b    = (const float*)d_in[5];
  const float* ln1g    = (const float*)d_in[6];
  const float* ln1b    = (const float*)d_in[7];
  const float* Wq      = (const float*)d_in[8];
  const float* Wkv     = (const float*)d_in[9];
  const float* ca_w    = (const float*)d_in[10];
  const float* ca_b    = (const float*)d_in[11];
  const float* ln2g    = (const float*)d_in[12];
  const float* ln2b    = (const float*)d_in[13];

  char* p = (char*)d_ws;
  auto take = [&](size_t bytes) { char* r = p; p += (bytes + 255) & ~(size_t)255; return r; };
  __bf16* tok_bf = (__bf16*)take(49152ull * 1024 * 2);
  __bf16* WqkvT  = (__bf16*)take(3072ull * 1024 * 2);
  __bf16* WsaT   = (__bf16*)take(1024ull * 1024 * 2);
  __bf16* WqT    = (__bf16*)take(1024ull * 1024 * 2);
  __bf16* WkvT   = (__bf16*)take(2048ull * 1024 * 2);
  __bf16* WcaT   = (__bf16*)take(1024ull * 1024 * 2);
  float*  qkv    = (float*) take(49152ull * 3072 * 4);
  __bf16* at1    = (__bf16*)take(49152ull * 1024 * 2);
  float*  saf    = (float*) take(49152ull * 1024 * 4);
  __bf16* tok2   = (__bf16*)take(49152ull * 1024 * 2);
  float*  t0     = (float*) take(16384ull * 1024 * 4);
  float*  qc     = (float*) take(16384ull * 1024 * 4);
  float*  kvc    = (float*) take(32768ull * 2048 * 4);
  __bf16* cain   = (__bf16*)take(16384ull * 1024 * 2);
  float*  caf    = (float*) take(16384ull * 1024 * 4);

  // 1) pack + convert
  pack_tokens<<<49152 * 1024 / 256, 256, 0, stream>>>(c_text, c_image, c_audio, tok_bf);
  transpose_w<<<(3072 * 1024) / 256, 256, 0, stream>>>(Wqkv, WqkvT, 1024, 3072);
  transpose_w<<<(1024 * 1024) / 256, 256, 0, stream>>>(sa_w, WsaT, 1024, 1024);
  transpose_w<<<(1024 * 1024) / 256, 256, 0, stream>>>(Wq, WqT, 1024, 1024);
  transpose_w<<<(2048 * 1024) / 256, 256, 0, stream>>>(Wkv, WkvT, 1024, 2048);
  transpose_w<<<(1024 * 1024) / 256, 256, 0, stream>>>(ca_w, WcaT, 1024, 1024);

  // 2) qkv = tokens @ Wqkv
  gemm_bf16_wmma<0><<<dim3(3072 / 128, 49152 / 128), 256, 0, stream>>>(
      tok_bf, WqkvT, qkv, nullptr, 49152, 3072, 1024, 1024);

  // 3) self-attention (seq 3)
  attn_self_kernel<<<NB * NHEAD / 8, 256, 0, stream>>>(qkv, at1);

  // 4) sa = attn @ sa_proj + b
  gemm_bf16_wmma<0><<<dim3(1024 / 128, 49152 / 128), 256, 0, stream>>>(
      at1, WsaT, saf, sa_b, 49152, 1024, 1024, 1024);

  // 5) tokens2 = LN1(tokens + sa)
  add_ln1_kernel<<<49152, 256, 0, stream>>>(c_text, c_image, c_audio, saf, ln1g, ln1b, tok2, t0);

  // 6) cq = t @ Wq   (A rows are token-0 rows, stride 3*1024)
  gemm_bf16_wmma<0><<<dim3(1024 / 128, 16384 / 128), 256, 0, stream>>>(
      tok2, WqT, qc, nullptr, 16384, 1024, 1024, 3072);

  // 7) kv = ia @ Wkv (A rows are token-1/2 rows, remapped in-kernel)
  gemm_bf16_wmma<1><<<dim3(2048 / 128, 32768 / 128), 256, 0, stream>>>(
      tok2, WkvT, kvc, nullptr, 32768, 2048, 1024, 0);

  // 8) cross-attention (1 query x 2 keys)
  attn_cross_kernel<<<NB * NHEAD / 8, 256, 0, stream>>>(qc, kvc, cain);

  // 9) ca = attn @ ca_proj + b
  gemm_bf16_wmma<0><<<dim3(1024 / 128, 16384 / 128), 256, 0, stream>>>(
      cain, WcaT, caf, ca_b, 16384, 1024, 1024, 1024);

  // 10) out = LN2(t + ca)
  final_ln_kernel<<<NB, 256, 0, stream>>>(t0, caf, ln2g, ln2b, (float*)d_out);
}
